// BatchRNN_15298673508812
// MI455X (gfx1250) — compile-verified, matrix operand order checked by reading, not run on hardware
//
#include <hip/hip_runtime.h>
#include <math.h>

#define BB 64
#define TT 1024
#define DD 512
#define HH 256
#define G4 (4*HH)       // 1024 gate columns
#define JW 32           // hidden units per workgroup (recurrent)
#define NWG (HH/JW)     // 8 workgroups per direction

typedef __attribute__((ext_vector_type(16))) __bf16 v16bf;
typedef __attribute__((ext_vector_type(8)))  __bf16 v8bf;
typedef __attribute__((ext_vector_type(8)))  float  v8f;

// ---------- WMMA fragment helpers ----------
// A fragment (16x32 bf16, row-major source, stride lda):
// lanes 0-15: row M=lane, K = {0..7, 16..23}; lanes 16-31: row M=lane-16, K = {8..15, 24..31}
__device__ __forceinline__ v16bf load_a_frag(const __bf16* A, int lda, int m0, int k0, int lane) {
    const __bf16* p = A + (size_t)(m0 + (lane & 15)) * lda + k0 + ((lane >> 4) << 3);
    v8bf lo = *(const v8bf*)p;
    v8bf hi = *(const v8bf*)(p + 16);
    return __builtin_shufflevector(lo, hi, 0,1,2,3,4,5,6,7,8,9,10,11,12,13,14,15);
}

// B fragments pre-packed: frag (kb, nb) stored as 32 lanes x 16 contiguous bf16
__device__ __forceinline__ v16bf load_b_frag_packed(const __bf16* Wpk, int ntiles, int kb, int nb, int lane) {
    return *(const v16bf*)(Wpk + (((size_t)kb * ntiles + nb) * 32 + (size_t)lane) * 16);
}

__device__ __forceinline__ float sigf(float x) { return 1.f / (1.f + __expf(-x)); }

// ---------- init: lengths, zero h double-buffers, zero barrier counters ----------
__global__ void init_kernel(const float* __restrict__ pad, int* lengths,
                            unsigned* hzf, unsigned* hzb, int* ctr) {
    int tid = threadIdx.x;
    if (tid < BB) {
        float s = 0.f;
        for (int t = 0; t < TT; ++t) s += pad[(size_t)tid * TT + t];
        lengths[tid] = TT - (int)(s + 0.5f);
    }
    const int HW = (2 * BB * HH * 2) / 4;   // uints per direction (double-buffered bf16 h)
    for (int i = tid; i < HW; i += 256) { hzf[i] = 0u; hzb[i] = 0u; }
    if (tid < 8) ctr[tid] = 0;
}

// ---------- batchnorm + pad mask -> bf16 x and bf16 x_rev (flip folded in) ----------
__global__ void bn_kernel(const float* __restrict__ in, const float* __restrict__ pad,
                          const float* __restrict__ scale, const float* __restrict__ bias,
                          const float* __restrict__ mean, const float* __restrict__ var,
                          const int* __restrict__ lengths,
                          __bf16* __restrict__ xf, __bf16* __restrict__ xr) {
    size_t idx = (size_t)blockIdx.x * 256 + threadIdx.x;
    int d = (int)(idx % DD);
    size_t bt = idx / DD;
    int t = (int)(bt % TT);
    int b = (int)(bt / TT);
    float inv = (1.f + scale[d]) * rsqrtf(var[d] + 1e-3f);
    float x = ((in[idx] - mean[d]) * inv + bias[d]) * (1.f - pad[bt]);
    __bf16 xbv = (__bf16)x;
    xf[idx] = xbv;
    int tr = (TT - 1 - t + lengths[b]) % TT;
    xr[((size_t)b * TT + tr) * DD + d] = xbv;
}

// ---------- pack a KxG4 fp32 weight matrix into bf16 WMMA B-fragment layout ----------
__global__ void pack_w_kernel(const float* __restrict__ W, __bf16* __restrict__ Wpk, int K) {
    int idx = blockIdx.x * 256 + threadIdx.x;
    if (idx >= K * G4) return;
    int k = idx / G4, n = idx % G4;
    int kb = k >> 5, kr = k & 31;
    int top = kr >> 4, q = kr & 15, h = q >> 3, r = q & 7;
    int e = top * 8 + r;
    int lane = (n & 15) + (h << 4);
    size_t dst = (((size_t)kb * (G4 / 16) + (n >> 4)) * 32 + lane) * 16 + e;
    Wpk[dst] = (__bf16)W[idx];
}

// ---------- input projection GEMM: xg = x @ Wx + b ----------
// Two-stage modulo-scheduled pipeline over k-blocks: no register rotation copies.
__global__ __launch_bounds__(256) void input_gemm_kernel(
        const __bf16* __restrict__ Xf, const __bf16* __restrict__ Xb,
        const __bf16* __restrict__ Wf, const __bf16* __restrict__ Wb,
        const float* __restrict__ bf_, const float* __restrict__ bb_,
        float* __restrict__ XGf, float* __restrict__ XGb) {
    const int dir = blockIdx.z;
    const __bf16* X = dir ? Xb : Xf;
    const __bf16* W = dir ? Wb : Wf;
    const float* bias = dir ? bb_ : bf_;
    float* XG = dir ? XGb : XGf;

    const int wave = threadIdx.x >> 5, lane = threadIdx.x & 31;
    const int m0 = blockIdx.x * 64 + (wave & 3) * 16;      // row in flattened (B*T)
    const int n0 = blockIdx.y * 128 + (wave >> 2) * 64;    // gate column
    const int nb0 = n0 >> 4;
    const int NKB = DD / 32;                                // 16 (even)

    v8f acc[4];
    const v8f z = {0.f,0.f,0.f,0.f,0.f,0.f,0.f,0.f};
#pragma unroll
    for (int nt = 0; nt < 4; ++nt) acc[nt] = z;

    // stage buffers
    v16bf a0, a1, b0[4], b1[4];
    a0 = load_a_frag(X, DD, m0, 0, lane);
#pragma unroll
    for (int nt = 0; nt < 4; ++nt) b0[nt] = load_b_frag_packed(W, G4 / 16, 0, nb0 + nt, lane);

    for (int kb = 0; kb < NKB; kb += 2) {
        // prefetch kb+1 into buf1, compute kb from buf0
        if (kb + 1 < NKB) {
            a1 = load_a_frag(X, DD, m0, (kb + 1) * 32, lane);
#pragma unroll
            for (int nt = 0; nt < 4; ++nt)
                b1[nt] = load_b_frag_packed(W, G4 / 16, kb + 1, nb0 + nt, lane);
        }
#pragma unroll
        for (int nt = 0; nt < 4; ++nt)
            acc[nt] = __builtin_amdgcn_wmma_f32_16x16x32_bf16(
                false, a0, false, b0[nt], (short)0, acc[nt], false, false);

        // prefetch kb+2 into buf0, compute kb+1 from buf1
        if (kb + 2 < NKB) {
            a0 = load_a_frag(X, DD, m0, (kb + 2) * 32, lane);
#pragma unroll
            for (int nt = 0; nt < 4; ++nt)
                b0[nt] = load_b_frag_packed(W, G4 / 16, kb + 2, nb0 + nt, lane);
        }
#pragma unroll
        for (int nt = 0; nt < 4; ++nt)
            acc[nt] = __builtin_amdgcn_wmma_f32_16x16x32_bf16(
                false, a1, false, b1[nt], (short)0, acc[nt], false, false);
    }

    const int mrow = m0 + ((lane & 16) ? 8 : 0);
#pragma unroll
    for (int nt = 0; nt < 4; ++nt) {
        const int col = n0 + nt * 16 + (lane & 15);
        const float bv = bias[col];
#pragma unroll
        for (int v = 0; v < 8; ++v)
            XG[(size_t)(mrow + v) * G4 + col] = acc[nt][v] + bv;
    }
}

// ---------- persistent recurrent scan (both directions; 8 WGs each) ----------
__global__ __launch_bounds__(256) void recurrent_kernel(
        const float* __restrict__ XGf, const float* __restrict__ XGb,
        const __bf16* __restrict__ Whf_pk, const __bf16* __restrict__ Whb_pk,
        __bf16* hf, __bf16* hb,
        const int* __restrict__ lengths,
        int* ctr_base, float* __restrict__ out) {
    const int dir = blockIdx.y;
    const int wg  = blockIdx.x;                 // owns hidden units [wg*JW, wg*JW+JW)
    const float* XG = dir ? XGb : XGf;
    const __bf16* Wpk = dir ? Whb_pk : Whf_pk;
    __bf16* hbuf = dir ? hb : hf;               // 2 * BB*HH bf16 (ping-pong)
    int* ctr = ctr_base + dir;
    const int j0 = wg * JW;

    const int wave = threadIdx.x >> 5, lane = threadIdx.x & 31;
    const int rt = wave & 3;                    // batch-row tile (16 rows)
    const int jt = wave >> 2;                   // which 16-wide j sub-tile

    // Stage this WG's Wh slice into LDS as ready-made B fragments:
    // 4 gates x 2 j-tiles x 8 k-blocks, 1 KB each = 64 KB
    __shared__ __align__(32) __bf16 lds_w[4 * 2 * 8 * 32 * 16];
    for (int i = threadIdx.x; i < 4 * 2 * 8 * 32; i += 256) {
        int f = i >> 5, l = i & 31;
        int gate = f >> 4, rem = f & 15, jt_ = rem >> 3, kb = rem & 7;
        int nb = (gate * HH + j0 + jt_ * 16) >> 4;
        const v16bf* src = (const v16bf*)(Wpk + (((size_t)kb * (G4 / 16) + nb) * 32 + l) * 16);
        *(v16bf*)(lds_w + (size_t)f * 512 + (size_t)l * 16) = *src;
    }
    __syncthreads();

    float c[8];
#pragma unroll
    for (int v = 0; v < 8; ++v) c[v] = 0.f;

    const int jcol  = j0 + jt * 16 + (lane & 15);
    const int bbase = rt * 16 + ((lane & 16) ? 8 : 0);
    int lenr[8];
#pragma unroll
    for (int v = 0; v < 8; ++v) lenr[v] = lengths[bbase + v];

    // seed pipeline: xg[:, 0, :]
    float xgbuf[4][8];
#pragma unroll
    for (int g = 0; g < 4; ++g)
#pragma unroll
        for (int v = 0; v < 8; ++v)
            xgbuf[g][v] = XG[((size_t)(bbase + v) * TT + 0) * G4 + g * HH + jcol];

    for (int t = 0; t < TT; ++t) {
        const __bf16* hcur = hbuf + (size_t)(t & 1) * (BB * HH);
        __bf16* hnext      = hbuf + (size_t)((t + 1) & 1) * (BB * HH);

        // accumulators seeded with xg[:, t, col] (bias already folded in)
        v8f acc[4];
#pragma unroll
        for (int g = 0; g < 4; ++g)
#pragma unroll
            for (int v = 0; v < 8; ++v)
                acc[g][v] = xgbuf[g][v];

        // gates += h_{t-1} @ Wh   (K = 256)
#pragma unroll
        for (int kb = 0; kb < HH / 32; ++kb) {
            v16bf a = load_a_frag(hcur, HH, rt * 16, kb * 32, lane);
#pragma unroll
            for (int g = 0; g < 4; ++g) {
                const v16bf bw = *(const v16bf*)(
                    lds_w + (size_t)(((g << 4) | (jt << 3) | kb)) * 512 + (size_t)lane * 16);
                acc[g] = __builtin_amdgcn_wmma_f32_16x16x32_bf16(
                    false, a, false, bw, (short)0, acc[g], false, false);
            }
        }

        // LSTM cell update, write h (bf16 state) and output (flip folded in for backward)
#pragma unroll
        for (int v = 0; v < 8; ++v) {
            float ig = sigf(acc[0][v]);
            float fg = sigf(acc[1][v]);
            float gg = tanhf(acc[2][v]);
            float og = sigf(acc[3][v]);
            c[v] = fg * c[v] + ig * gg;
            float h = og * tanhf(c[v]);
            const int b = bbase + v;
            hnext[(size_t)b * HH + jcol] = (__bf16)h;
            const int tpos = dir ? ((TT - 1 - t + lenr[v]) % TT) : t;
            out[((size_t)b * TT + tpos) * (2 * HH) + dir * HH + jcol] = h;
        }

        // prefetch xg seeds for t+1 BEFORE the barrier spin: these loads do not
        // depend on the barrier, so their latency hides under barrier latency
        if (t + 1 < TT) {
#pragma unroll
            for (int g = 0; g < 4; ++g)
#pragma unroll
                for (int v = 0; v < 8; ++v)
                    xgbuf[g][v] = XG[((size_t)(bbase + v) * TT + (t + 1)) * G4 + g * HH + jcol];
            // warm L2 a few steps ahead of the 256 MB/dir xg stream
            if (t + 8 < TT)
                __builtin_prefetch(&XG[((size_t)bbase * TT + (t + 8)) * G4 + jcol], 0, 0);
        }

        // inter-workgroup barrier (per direction): generation counter, never reset
        __syncthreads();
        if (threadIdx.x == 0) {
            __threadfence();
            atomicAdd(ctr, 1);
            const int target = NWG * (t + 1);
            while (__hip_atomic_load(ctr, __ATOMIC_ACQUIRE, __HIP_MEMORY_SCOPE_AGENT) < target) {
                __builtin_amdgcn_s_sleep(1);
            }
        }
        __syncthreads();
    }
}

extern "C" void kernel_launch(void* const* d_in, const int* in_sizes, int n_in,
                              void* d_out, int out_size, void* d_ws, size_t ws_size,
                              hipStream_t stream) {
    const float* inputs   = (const float*)d_in[0];
    const float* pad      = (const float*)d_in[1];
    const float* bn_scale = (const float*)d_in[2];
    const float* bn_bias  = (const float*)d_in[3];
    const float* bn_mean  = (const float*)d_in[4];
    const float* bn_var   = (const float*)d_in[5];
    const float* Wx_f     = (const float*)d_in[6];
    const float* Wh_f     = (const float*)d_in[7];
    const float* b_f      = (const float*)d_in[8];
    const float* Wx_b     = (const float*)d_in[9];
    const float* Wh_b     = (const float*)d_in[10];
    const float* b_b      = (const float*)d_in[11];
    float* out = (float*)d_out;

    char* ws = (char*)d_ws;
    size_t off = 0;
    auto take = [&](size_t bytes) -> char* {
        char* p = ws + off;
        off = (off + bytes + 255) & ~(size_t)255;
        return p;
    };
    __bf16* x_bf    = (__bf16*)take((size_t)BB * TT * DD * 2);
    __bf16* xr_bf   = (__bf16*)take((size_t)BB * TT * DD * 2);
    __bf16* wxf_pk  = (__bf16*)take((size_t)DD * G4 * 2);
    __bf16* wxb_pk  = (__bf16*)take((size_t)DD * G4 * 2);
    __bf16* whf_pk  = (__bf16*)take((size_t)HH * G4 * 2);
    __bf16* whb_pk  = (__bf16*)take((size_t)HH * G4 * 2);
    float*  xg_f    = (float*)take((size_t)BB * TT * G4 * 4);
    float*  xg_b    = (float*)take((size_t)BB * TT * G4 * 4);
    __bf16* h_f     = (__bf16*)take((size_t)2 * BB * HH * 2);
    __bf16* h_b     = (__bf16*)take((size_t)2 * BB * HH * 2);
    int*    lengths = (int*)take(BB * 4);
    int*    ctr     = (int*)take(64);

    init_kernel<<<1, 256, 0, stream>>>(pad, lengths, (unsigned*)h_f, (unsigned*)h_b, ctr);
    bn_kernel<<<(BB * TT * DD) / 256, 256, 0, stream>>>(
        inputs, pad, bn_scale, bn_bias, bn_mean, bn_var, lengths, x_bf, xr_bf);
    pack_w_kernel<<<(DD * G4) / 256, 256, 0, stream>>>(Wx_f, wxf_pk, DD);
    pack_w_kernel<<<(DD * G4) / 256, 256, 0, stream>>>(Wx_b, wxb_pk, DD);
    pack_w_kernel<<<(HH * G4) / 256, 256, 0, stream>>>(Wh_f, whf_pk, HH);
    pack_w_kernel<<<(HH * G4) / 256, 256, 0, stream>>>(Wh_b, whb_pk, HH);
    input_gemm_kernel<<<dim3((BB * TT) / 64, G4 / 128, 2), 256, 0, stream>>>(
        x_bf, xr_bf, wxf_pk, wxb_pk, b_f, b_b, xg_f, xg_b);
    recurrent_kernel<<<dim3(NWG, 2), 256, 0, stream>>>(
        xg_f, xg_b, whf_pk, whb_pk, h_f, h_b, lengths, ctr, out);
}
